// MoETransformerLayer_78795470012600
// MI455X (gfx1250) — matrix operand order repaired and checked by reading, not compiled, not run on hardware
//
#include <hip/hip_runtime.h>
#include <hip/hip_bf16.h>

// ---------------------------------------------------------------------------
// MoE transformer layer for gfx1250 (CDNA5, wave32, WMMA bf16 + TDM staging).
// ---------------------------------------------------------------------------

typedef __bf16 bf16;
typedef __attribute__((ext_vector_type(16))) __bf16 bf16x16;
typedef __attribute__((ext_vector_type(8)))  float  floatx8;
typedef __attribute__((ext_vector_type(4)))  unsigned int uint32x4;
typedef __attribute__((ext_vector_type(8)))  int int32x8;
typedef __attribute__((ext_vector_type(4)))  int int32x4;

union FragBF { bf16x16 v; unsigned int u[8]; };

constexpr int kB  = 8;
constexpr int kS  = 1024;
constexpr int kD  = 1024;
constexpr int kH  = 16;
constexpr int kHD = 64;
constexpr int kM  = kB * kS;        // 8192 tokens
constexpr int kSharedH = 4096;
constexpr int kRoutedH = 1024;
constexpr int kNExp = 8;

// TDM availability (this toolchain: 6-arg builtin). Fallback: manual loads.
#if defined(__has_builtin)
#  if __has_builtin(__builtin_amdgcn_tensor_load_to_lds) && \
      __has_builtin(__builtin_amdgcn_s_wait_tensorcnt)
#    define USE_TDM 1
#  endif
#endif
#ifndef USE_TDM
#  define USE_TDM 0
#endif

__device__ __forceinline__ bf16  f2bf(float f) { return (bf16)f; }
__device__ __forceinline__ float bf2f(bf16 h)  { return (float)h; }
__device__ __forceinline__ unsigned int bfbits(bf16 h) {
  unsigned short u; __builtin_memcpy(&u, &h, 2); return (unsigned int)u;
}
// LDS byte address of a shared-memory pointer (generic LDS addr: low 32 bits)
__device__ __forceinline__ unsigned lds_addr32(const void* p) {
  return (unsigned)(unsigned long long)p;
}

__device__ __forceinline__ floatx8 wmma_bf16(const FragBF& a, const FragBF& b, floatx8 c) {
  return __builtin_amdgcn_wmma_f32_16x16x32_bf16(false, a.v, false, b.v, (short)0, c,
                                                 false, false);
}

// A-matrix fragment (16x32 bf16, M x K), tile row-major [16][32].
__device__ __forceinline__ void load_fragA(FragBF& f, const bf16* tile, int lane) {
  const unsigned int* row = (const unsigned int*)(tile + (size_t)(lane & 15) * 32);
  const int b0 = (lane < 16) ? 0 : 8;   // K-pair index base
  f.u[0] = row[b0 + 0]; f.u[1] = row[b0 + 1]; f.u[2] = row[b0 + 2]; f.u[3] = row[b0 + 3];
  f.u[4] = row[b0 + 4]; f.u[5] = row[b0 + 5]; f.u[6] = row[b0 + 6]; f.u[7] = row[b0 + 7];
}

// B-matrix fragment (32x16 bf16, K x N) from an n-major LDS tile [16 n][32 k].
__device__ __forceinline__ void load_fragB_nk(FragBF& f, const bf16* tile, int lane) {
  const unsigned int* row = (const unsigned int*)(tile + (size_t)(lane & 15) * 32);
  const int b0 = (lane < 16) ? 0 : 8;
#pragma unroll
  for (int v = 0; v < 8; ++v) f.u[v] = row[b0 + v];
}

#if USE_TDM
// Issue one TDM 2-D tile load: 128 rows x 32 bf16, row stride = strideK elements.
// D# per ISA 8.3/8.4: group0 {count, lds_addr, global_addr, type=2},
// group1 {data_size=2B, dims, strides}, groups 2/3 zero (<=2D tensor).
__device__ __forceinline__ void tdm_load_tile_128x32(const bf16* gsrc, int strideK,
                                                     unsigned lds_addr) {
  const unsigned long long ga = (unsigned long long)(const void*)gsrc;
  uint32x4 g0 = { 1u,                                   // count=1, user desc
                  lds_addr,                             // lds byte address
                  (unsigned)ga,                         // global addr [31:0]
                  (unsigned)(ga >> 32) | 0x80000000u }; // addr[56:32] | type=2
  int32x8 g1 = { 0x00010000,            // data_size=1 (2 bytes), no mask/flags
                 0x00200000,            // tensor_dim0 = 32  (bits 79:48 lo)
                 0x00800000,            // tensor_dim1 = 128 (bits 111:80 lo)
                 0x00200000,            // tile_dim0 = 32    (bits 127:112)
                 128,                   // tile_dim1 = 128, tile_dim2 = 0
                 strideK,               // tensor_dim0_stride[31:0] (elements)
                 0, 0 };
  int32x4 z4 = { 0, 0, 0, 0 };
  int32x8 z8 = { 0, 0, 0, 0, 0, 0, 0, 0 };
  __builtin_amdgcn_tensor_load_to_lds(g0, g1, z4, z4, z8, 0);
}
#endif

// ---------------------------------------------------------------------------
// fp32 -> bf16 conversion
// ---------------------------------------------------------------------------
__global__ void cvt_f32_bf16_kernel(const float* __restrict__ in, bf16* __restrict__ out,
                                    long long n) {
  long long i = (long long)blockIdx.x * blockDim.x + threadIdx.x;
  long long stride = (long long)gridDim.x * blockDim.x;
  for (; i < n; i += stride) out[i] = f2bf(in[i]);
}

// ---------------------------------------------------------------------------
// RMSNorm: fp32 row -> bf16 row
// ---------------------------------------------------------------------------
__global__ __launch_bounds__(256) void rmsnorm_kernel(const float* __restrict__ x,
                                                      const float* __restrict__ w,
                                                      bf16* __restrict__ out) {
  __shared__ float red[8];
  const int row = blockIdx.x;
  const float* xr = x + (size_t)row * kD;
  float ss = 0.f;
  for (int i = threadIdx.x; i < kD; i += 256) { float v = xr[i]; ss += v * v; }
#pragma unroll
  for (int o = 16; o > 0; o >>= 1) ss += __shfl_xor(ss, o, 32);
  if ((threadIdx.x & 31) == 0) red[threadIdx.x >> 5] = ss;
  __syncthreads();
  float tot = 0.f;
#pragma unroll
  for (int i = 0; i < 8; ++i) tot += red[i];
  const float inv = rsqrtf(tot / (float)kD + 1e-5f);
  for (int i = threadIdx.x; i < kD; i += 256)
    out[(size_t)row * kD + i] = f2bf(xr[i] * inv * w[i]);
}

// ---------------------------------------------------------------------------
// Generic WMMA GEMM:  C[M,N] = A[M,K](bf16,row-major) * W[N,K](bf16)^T
// Block: 256 threads (8 waves). Tile 128(M) x 128(N), K-step 32.
// Waves 4x2: each wave computes 32(M) x 64(N) = 8 WMMA per K-step.
// TDM double-buffered LDS staging when available.
// ---------------------------------------------------------------------------
struct GP {
  const bf16 *A, *W, *W2;
  const float *bias, *bias2, *src, *gamma, *routew;
  float* outF;
  bf16*  outH;
  int Mdim, Ndim, Kdim, expert;
};

template <int MODE>
__global__ __launch_bounds__(256) void gemm_kernel(GP p) {
  __shared__ alignas(16) bf16 As[2 * 128 * 32];
  __shared__ alignas(16) bf16 Bs[2 * 128 * 32];
  __shared__ alignas(16) bf16 Bs2[2 * 128 * 32];

  const int tid  = threadIdx.x;
  const int n0   = blockIdx.x * 128;
  const int m0   = blockIdx.y * 128;
  const int wave = tid >> 5;
  const int lane = tid & 31;
  const int wrow = wave >> 1;   // 0..3 -> M offset 32*wrow
  const int wcol = wave & 1;    // 0..1 -> N offset 64*wcol
  const int K    = p.Kdim;

  floatx8 acc[2][4]  = {};
  floatx8 acc2[2][4] = {};

  auto do_compute = [&](int buf) {
    const bf16* Ab  = As  + buf * 4096;
    const bf16* Bb  = Bs  + buf * 4096;
    const bf16* Bb2 = Bs2 + buf * 4096;
    FragBF fa0, fa1;
    load_fragA(fa0, Ab + (size_t)(wrow * 32) * 32, lane);
    load_fragA(fa1, Ab + (size_t)(wrow * 32 + 16) * 32, lane);
#pragma unroll
    for (int nt = 0; nt < 4; ++nt) {
      FragBF fb;
      load_fragB_nk(fb, Bb + (size_t)(wcol * 64 + nt * 16) * 32, lane);
      acc[0][nt] = wmma_bf16(fa0, fb, acc[0][nt]);
      acc[1][nt] = wmma_bf16(fa1, fb, acc[1][nt]);
      if (MODE == 2) {
        FragBF fb2;
        load_fragB_nk(fb2, Bb2 + (size_t)(wcol * 64 + nt * 16) * 32, lane);
        acc2[0][nt] = wmma_bf16(fa0, fb2, acc2[0][nt]);
        acc2[1][nt] = wmma_bf16(fa1, fb2, acc2[1][nt]);
      }
    }
  };

#if USE_TDM
  constexpr int NTDM = (MODE == 2) ? 3 : 2;   // tiles in flight per issue
  auto issue = [&](int k0, int buf) {
    if (wave == 0) {   // wave-uniform branch: one DMA set per block
      tdm_load_tile_128x32(p.A + (size_t)m0 * K + k0, K,
                           lds_addr32(As + buf * 4096));
      tdm_load_tile_128x32(p.W + (size_t)n0 * K + k0, K,
                           lds_addr32(Bs + buf * 4096));
      if (MODE == 2)
        tdm_load_tile_128x32(p.W2 + (size_t)n0 * K + k0, K,
                             lds_addr32(Bs2 + buf * 4096));
    }
  };
  issue(0, 0);
  for (int k0 = 0; k0 < K; k0 += 32) {
    const int buf = (k0 >> 5) & 1;
    const bool more = (k0 + 32 < K);
    if (more) issue(k0 + 32, buf ^ 1);   // overlap next DMA with compute
    if (wave == 0) {
      if (more) __builtin_amdgcn_s_wait_tensorcnt(NTDM);  // current tiles done
      else      __builtin_amdgcn_s_wait_tensorcnt(0);
    }
    __syncthreads();      // all waves see completed LDS tiles
    do_compute(buf);
    __syncthreads();      // done reading before next DMA overwrites
  }
#else
  for (int k0 = 0; k0 < K; k0 += 32) {
#pragma unroll
    for (int i = 0; i < 2; ++i) {
      int idx = tid + i * 256;
      int r = idx >> 2, c = (idx & 3) * 8;
      *(uint4*)&As[r * 32 + c] = *(const uint4*)(p.A + (size_t)(m0 + r) * K + k0 + c);
      *(uint4*)&Bs[r * 32 + c] = *(const uint4*)(p.W + (size_t)(n0 + r) * K + k0 + c);
      if (MODE == 2)
        *(uint4*)&Bs2[r * 32 + c] = *(const uint4*)(p.W2 + (size_t)(n0 + r) * K + k0 + c);
    }
    if (k0 + 32 < K) {
      __builtin_prefetch(p.A + (size_t)(m0 + (tid >> 1)) * K + k0 + 32, 0, 3);
      __builtin_prefetch(p.W + (size_t)(n0 + (tid >> 1)) * K + k0 + 32, 0, 3);
    }
    __syncthreads();
    do_compute(0);
    __syncthreads();
  }
#endif

  // ---- epilogue ----
  const int nlo = lane & 15;
  const int mhi = (lane < 16) ? 0 : 8;   // C layout: VGPR r -> row r / r+8
#pragma unroll
  for (int mt = 0; mt < 2; ++mt) {
#pragma unroll
    for (int nt = 0; nt < 4; ++nt) {
#pragma unroll
      for (int r = 0; r < 8; ++r) {
        const int mm = m0 + wrow * 32 + mt * 16 + mhi + r;
        const int nn = n0 + wcol * 64 + nt * 16 + nlo;
        float val = acc[mt][nt][r];
        if (MODE == 0) {
          // QKV: scatter to [3][B][H][S][64] bf16 (+bias)
          val += p.bias[nn];
          const int which = nn >> 10, rr = nn & 1023;
          const int h = rr >> 6, dh = rr & 63;
          const int b = mm >> 10, s = mm & 1023;
          p.outH[((((size_t)which * kB + b) * kH + h) * kS + s) * kHD + dh] = f2bf(val);
        } else if (MODE == 1) {
          // out-proj fused residual: x1 = src + gamma1*(acc+bias)
          val += p.bias[nn];
          const size_t idx = (size_t)mm * kD + nn;
          p.outF[idx] = p.src[idx] + p.gamma[nn] * val;
        } else if (MODE == 2) {
          // fused SwiGLU: h = silu(gate) * up  -> bf16 [M,N]
          const float g = val + p.bias[nn];
          const float u = acc2[mt][nt][r] + p.bias2[nn];
          const float sg = g * (1.0f / (1.0f + __expf(-g)));
          p.outH[(size_t)mm * p.Ndim + nn] = f2bf(sg * u);
        } else if (MODE == 3) {
          // shared down-proj -> ff fp32 (initializes accumulator)
          p.outF[(size_t)mm * p.Ndim + nn] = val + p.bias[nn];
        } else {
          // routed down-proj: ff += routing_w[b,e] * (acc+bias)
          const float w = p.routew[(mm >> 10) * kNExp + p.expert];
          atomicAdd(&p.outF[(size_t)mm * p.Ndim + nn], w * (val + p.bias[nn]));
        }
      }
    }
  }
}

// ---------------------------------------------------------------------------
// RoPE-4D on q or k: read bf16 qkv slice, write bf16 [B,H,S,64].
// ---------------------------------------------------------------------------
__global__ __launch_bounds__(32) void rope_kernel(const bf16* __restrict__ qkvh,
                                                  const float* __restrict__ cent,
                                                  bf16* __restrict__ dst, int which) {
  const int gid = blockIdx.x;              // b*H*S + h*S + s
  const int s = gid % kS;
  const int h = (gid / kS) % kH;
  const int b = gid / (kS * kH);
  const int lane = threadIdx.x;
  const int a = lane >> 3, i = lane & 7;

  const bf16* src = qkvh + (((size_t)which * kB + b) * kH + h) * (size_t)kS * kHD
                         + (size_t)s * kHD;
  const float scale = (a == 3) ? 0.2f : 1.0f;
  const float pos  = cent[((size_t)b * kS + s) * 4 + a] * scale;
  const float invf = __powf(10000.0f, -(float)i * 0.125f);
  const float ang  = pos * invf;
  const float c = __cosf(ang), sn = __sinf(ang);
  const float x1 = bf2f(src[a * 16 + i]);
  const float x2 = bf2f(src[a * 16 + 8 + i]);
  bf16* d = dst + (((size_t)b * kH + h) * kS + s) * kHD;
  d[a * 16 + i]     = f2bf(x1 * c - x2 * sn);
  d[a * 16 + 8 + i] = f2bf(x1 * sn + x2 * c);
}

// ---------------------------------------------------------------------------
// Attention: one wave per (head, 16-query tile).
// ---------------------------------------------------------------------------
__global__ __launch_bounds__(32) void attn_kernel(const bf16* __restrict__ qm,
                                                  const bf16* __restrict__ km,
                                                  const bf16* __restrict__ vm,
                                                  bf16* __restrict__ outT) {
  __shared__ bf16 s_sc[16 * kS];               // 32 KB scores -> probs
  __shared__ alignas(16) bf16 s_v[32 * kHD];   // 4 KB V tile

  const int lane = threadIdx.x;
  const int qt = blockIdx.x;
  const int bh = blockIdx.y;
  const int b = bh / kH, h = bh % kH;
  const bf16* qb = qm + (size_t)bh * kS * kHD;
  const bf16* kb = km + (size_t)bh * kS * kHD;
  const bf16* vb = vm + (size_t)bh * kS * kHD;
  const int m0 = qt * 16;
  const int nlo = lane & 15;
  const int mhi = (lane < 16) ? 0 : 8;

  FragBF aq[2];
  {
    const unsigned int* rowp = (const unsigned int*)(qb + (size_t)(m0 + nlo) * kHD);
    const int b0 = (lane < 16) ? 0 : 8;
#pragma unroll
    for (int kk = 0; kk < 2; ++kk) {
      aq[kk].u[0] = rowp[kk * 16 + b0 + 0];
      aq[kk].u[1] = rowp[kk * 16 + b0 + 1];
      aq[kk].u[2] = rowp[kk * 16 + b0 + 2];
      aq[kk].u[3] = rowp[kk * 16 + b0 + 3];
      aq[kk].u[4] = rowp[kk * 16 + b0 + 4];
      aq[kk].u[5] = rowp[kk * 16 + b0 + 5];
      aq[kk].u[6] = rowp[kk * 16 + b0 + 6];
      aq[kk].u[7] = rowp[kk * 16 + b0 + 7];
    }
  }

  // scores = q.k^T / sqrt(64)
  for (int j = 0; j < kS / 16; ++j) {
    floatx8 acc = {};
    const unsigned int* rowp = (const unsigned int*)(kb + (size_t)(j * 16 + nlo) * kHD);
    const int b0 = (lane < 16) ? 0 : 8;
#pragma unroll
    for (int kk = 0; kk < 2; ++kk) {
      FragBF bk;
#pragma unroll
      for (int v = 0; v < 8; ++v) bk.u[v] = rowp[kk * 16 + b0 + v];
      acc = wmma_bf16(aq[kk], bk, acc);
    }
#pragma unroll
    for (int r = 0; r < 8; ++r)
      s_sc[(size_t)(mhi + r) * kS + j * 16 + nlo] = f2bf(acc[r] * 0.125f);
  }
  __syncthreads();

  // softmax (pad mask is all-false)
  {
    bf16* rowsc = s_sc + (size_t)nlo * kS + (lane >> 4) * 512;
    float mx = -3.0e38f;
    for (int i = 0; i < 512; ++i) mx = fmaxf(mx, bf2f(rowsc[i]));
    mx = fmaxf(mx, __shfl_xor(mx, 16, 32));
    float sum = 0.f;
    for (int i = 0; i < 512; ++i) {
      float e = __expf(bf2f(rowsc[i]) - mx);
      sum += e;
      rowsc[i] = f2bf(e);
    }
    sum += __shfl_xor(sum, 16, 32);
    const float inv = 1.f / sum;
    for (int i = 0; i < 512; ++i) rowsc[i] = f2bf(bf2f(rowsc[i]) * inv);
  }
  __syncthreads();

  // O = P @ V
  floatx8 accO[4] = {};
  for (int kk = 0; kk < kS / 32; ++kk) {
    const uint4* vr = (const uint4*)(vb + (size_t)(kk * 32 + lane) * kHD);
    uint4* dv = (uint4*)(s_v + (size_t)lane * kHD);
#pragma unroll
    for (int t = 0; t < 8; ++t) dv[t] = vr[t];
    __syncthreads();

    FragBF ap;
    {
      const unsigned int* rowp = (const unsigned int*)(s_sc + (size_t)nlo * kS + kk * 32);
      const int b0 = (lane < 16) ? 0 : 8;
      ap.u[0] = rowp[b0 + 0]; ap.u[1] = rowp[b0 + 1];
      ap.u[2] = rowp[b0 + 2]; ap.u[3] = rowp[b0 + 3];
      ap.u[4] = rowp[b0 + 4]; ap.u[5] = rowp[b0 + 5];
      ap.u[6] = rowp[b0 + 6]; ap.u[7] = rowp[b0 + 7];
    }
    const int kb0 = (lane < 16) ? 0 : 16;
#pragma unroll
    for (int nt = 0; nt < 4; ++nt) {
      FragBF bv;
      const int n = nt * 16 + nlo;
#pragma unroll
      for (int v = 0; v < 8; ++v) {
        unsigned int lo = bfbits(s_v[(size_t)(kb0 + 2 * v) * kHD + n]);
        unsigned int hi = bfbits(s_v[(size_t)(kb0 + 2 * v + 1) * kHD + n]);
        bv.u[v] = lo | (hi << 16);
      }
      accO[nt] = wmma_bf16(ap, bv, accO[nt]);
    }
    __syncthreads();
  }

#pragma unroll
  for (int nt = 0; nt < 4; ++nt)
#pragma unroll
    for (int r = 0; r < 8; ++r) {
      const int s = m0 + mhi + r;
      outT[(((size_t)b * kS + s) * kH + h) * kHD + nt * 16 + nlo] = f2bf(accO[nt][r]);
    }
}

// ---------------------------------------------------------------------------
// Final residual: out = x1 + gamma2 * ff
// ---------------------------------------------------------------------------
__global__ void final_residual_kernel(const float* __restrict__ x1,
                                      const float* __restrict__ ff,
                                      const float* __restrict__ gamma2,
                                      float* __restrict__ out, long long n) {
  long long i = (long long)blockIdx.x * blockDim.x + threadIdx.x;
  long long stride = (long long)gridDim.x * blockDim.x;
  for (; i < n; i += stride) out[i] = x1[i] + gamma2[i & (kD - 1)] * ff[i];
}

// ---------------------------------------------------------------------------
// Host-side launch sequence
// ---------------------------------------------------------------------------
extern "C" void kernel_launch(void* const* d_in, const int* in_sizes, int n_in,
                              void* d_out, int out_size, void* d_ws, size_t ws_size,
                              hipStream_t stream) {
  (void)in_sizes; (void)n_in; (void)out_size; (void)ws_size;
  const float* src    = (const float*)d_in[0];
  const float* cent   = (const float*)d_in[1];
  const float* routew = (const float*)d_in[2];
  const float* qkv_w = (const float*)d_in[4];
  const float* qkv_b = (const float*)d_in[5];
  const float* out_w = (const float*)d_in[6];
  const float* out_b = (const float*)d_in[7];
  const float* n1w   = (const float*)d_in[8];
  const float* n2w   = (const float*)d_in[9];
  const float* g1    = (const float*)d_in[10];
  const float* g2    = (const float*)d_in[11];
  const float* sh_wg = (const float*)d_in[12];
  const float* sh_bg = (const float*)d_in[13];
  const float* sh_wu = (const float*)d_in[14];
  const float* sh_bu = (const float*)d_in[15];
  const float* sh_wd = (const float*)d_in[16];
  const float* sh_bd = (const float*)d_in[17];
  const float* r_wg  = (const float*)d_in[18];
  const float* r_bg  = (const float*)d_in[19];
  const float* r_wu  = (const float*)d_in[20];
  const float* r_bu  = (const float*)d_in[21];
  const float* r_wd  = (const float*)d_in[22];
  const float* r_bd  = (const float*)d_in[23];
  float* out = (float*)d_out;

  char* ws = (char*)d_ws;
  size_t off = 0;
  auto alloc = [&](size_t bytes) -> void* {
    void* p = ws + off;
    off += (bytes + 255) & ~(size_t)255;
    return p;
  };
  bf16* wq   = (bf16*)alloc((size_t)3 * kD * kD * 2);
  bf16* wo   = (bf16*)alloc((size_t)kD * kD * 2);
  bf16* wsg  = (bf16*)alloc((size_t)kSharedH * kD * 2);
  bf16* wsu  = (bf16*)alloc((size_t)kSharedH * kD * 2);
  bf16* wsd  = (bf16*)alloc((size_t)kD * kSharedH * 2);
  bf16* wrg  = (bf16*)alloc((size_t)kNExp * kRoutedH * kD * 2);
  bf16* wru  = (bf16*)alloc((size_t)kNExp * kRoutedH * kD * 2);
  bf16* wrd  = (bf16*)alloc((size_t)kNExp * kD * kRoutedH * 2);
  bf16* xn   = (bf16*)alloc((size_t)kM * kD * 2);
  bf16* qkvh = (bf16*)alloc((size_t)3 * kM * kD * 2);
  bf16* qh   = (bf16*)alloc((size_t)kM * kD * 2);
  bf16* kh   = (bf16*)alloc((size_t)kM * kD * 2);
  bf16* aoT  = (bf16*)alloc((size_t)kM * kD * 2);
  float* x1  = (float*)alloc((size_t)kM * kD * 4);
  bf16* xn2  = (bf16*)alloc((size_t)kM * kD * 2);
  bf16* hsh  = (bf16*)alloc((size_t)kM * kSharedH * 2);
  float* ff  = (float*)alloc((size_t)kM * kD * 4);
  bf16* hr   = (bf16*)alloc((size_t)kM * kRoutedH * 2);

  const bf16* vh = qkvh + (size_t)2 * kM * kD;  // v slice (no rope)

  auto cvt = [&](const float* in, bf16* o, long long n) {
    cvt_f32_bf16_kernel<<<2048, 256, 0, stream>>>(in, o, n);
  };

  cvt(qkv_w, wq,  (long long)3 * kD * kD);
  cvt(out_w, wo,  (long long)kD * kD);
  cvt(sh_wg, wsg, (long long)kSharedH * kD);
  cvt(sh_wu, wsu, (long long)kSharedH * kD);
  cvt(sh_wd, wsd, (long long)kD * kSharedH);
  cvt(r_wg,  wrg, (long long)kNExp * kRoutedH * kD);
  cvt(r_wu,  wru, (long long)kNExp * kRoutedH * kD);
  cvt(r_wd,  wrd, (long long)kNExp * kD * kRoutedH);

  rmsnorm_kernel<<<kM, 256, 0, stream>>>(src, n1w, xn);

  {
    GP p{};
    p.A = xn; p.W = wq; p.bias = qkv_b; p.outH = qkvh;
    p.Mdim = kM; p.Ndim = 3 * kD; p.Kdim = kD;
    gemm_kernel<0><<<dim3((3 * kD) / 128, kM / 128), 256, 0, stream>>>(p);
  }

  rope_kernel<<<kB * kH * kS, 32, 0, stream>>>(qkvh, cent, qh, 0);
  rope_kernel<<<kB * kH * kS, 32, 0, stream>>>(qkvh, cent, kh, 1);

  attn_kernel<<<dim3(kS / 16, kB * kH), 32, 0, stream>>>(qh, kh, vh, aoT);

  {
    GP p{};
    p.A = aoT; p.W = wo; p.bias = out_b; p.src = src; p.gamma = g1; p.outF = x1;
    p.Mdim = kM; p.Ndim = kD; p.Kdim = kD;
    gemm_kernel<1><<<dim3(kD / 128, kM / 128), 256, 0, stream>>>(p);
  }

  rmsnorm_kernel<<<kM, 256, 0, stream>>>(x1, n2w, xn2);

  {
    GP p{};
    p.A = xn2; p.W = wsg; p.W2 = wsu; p.bias = sh_bg; p.bias2 = sh_bu; p.outH = hsh;
    p.Mdim = kM; p.Ndim = kSharedH; p.Kdim = kD;
    gemm_kernel<2><<<dim3(kSharedH / 128, kM / 128), 256, 0, stream>>>(p);
  }
  {
    GP p{};
    p.A = hsh; p.W = wsd; p.bias = sh_bd; p.outF = ff;
    p.Mdim = kM; p.Ndim = kD; p.Kdim = kSharedH;
    gemm_kernel<3><<<dim3(kD / 128, kM / 128), 256, 0, stream>>>(p);
  }

  for (int e = 0; e < kNExp; ++e) {
    {
      GP p{};
      p.A = xn2;
      p.W  = wrg + (size_t)e * kRoutedH * kD;
      p.W2 = wru + (size_t)e * kRoutedH * kD;
      p.bias = r_bg + (size_t)e * kRoutedH;
      p.bias2 = r_bu + (size_t)e * kRoutedH;
      p.outH = hr;
      p.Mdim = kM; p.Ndim = kRoutedH; p.Kdim = kD;
      gemm_kernel<2><<<dim3(kRoutedH / 128, kM / 128), 256, 0, stream>>>(p);
    }
    {
      GP p{};
      p.A = hr;
      p.W = wrd + (size_t)e * kD * kRoutedH;
      p.bias = r_bd + (size_t)e * kD;
      p.outF = ff; p.routew = routew; p.expert = e;
      p.Mdim = kM; p.Ndim = kD; p.Kdim = kRoutedH;
      gemm_kernel<4><<<dim3(kD / 128, kM / 128), 256, 0, stream>>>(p);
    }
  }

  final_residual_kernel<<<8192, 256, 0, stream>>>(x1, ff, g2, out,
                                                  (long long)kM * kD);
}
